// AssetSimilarityNetwork_53609781789044
// MI455X (gfx1250) — matrix operand order repaired and checked by reading, not compiled, not run on hardware
//
#include <hip/hip_runtime.h>
#include <math.h>

typedef __attribute__((ext_vector_type(16))) _Float16 v16h;
typedef __attribute__((ext_vector_type(8)))  float    v8f;

#define NN 2048
#define ICHUNK 64   // i values per wave -> 128 j-tiles x 32 chunks = 4096 waves

// ---------------------------------------------------------------------------
// Kernel 1: tiny precompute. proj = asset @ Wp^T + bp; pa = proj @ W1[:, :32]^T;
// pbb = proj @ W1[:, 32:]^T + b1.  2048 threads, ~8.4M MACs total -> negligible.
// ---------------------------------------------------------------------------
__global__ __launch_bounds__(256)
void precompute_papb(const float* __restrict__ asset,  // 2048 x 64
                     const float* __restrict__ Wp,     // 32 x 64
                     const float* __restrict__ bp,     // 32
                     const float* __restrict__ W1,     // 32 x 64
                     const float* __restrict__ b1,     // 32
                     float* __restrict__ pa,           // 2048 x 32
                     float* __restrict__ pbb)          // 2048 x 32  (pb + b1)
{
    int n = blockIdx.x * blockDim.x + threadIdx.x;
    if (n >= NN) return;
    float feat[64];
    const float* arow = asset + n * 64;
    #pragma unroll
    for (int f = 0; f < 64; ++f) feat[f] = arow[f];

    float proj[32];
    for (int p = 0; p < 32; ++p) {
        float acc = bp[p];
        const float* wrow = Wp + p * 64;
        #pragma unroll
        for (int f = 0; f < 64; ++f) acc = fmaf(feat[f], wrow[f], acc);
        proj[p] = acc;
    }
    for (int a = 0; a < 32; ++a) {
        const float* wrow = W1 + a * 64;   // Wa = W1[:, :32], Wb = W1[:, 32:]
        float s0 = 0.f, s1 = 0.f;
        #pragma unroll
        for (int p = 0; p < 32; ++p) {
            s0 = fmaf(proj[p], wrow[p],      s0);
            s1 = fmaf(proj[p], wrow[32 + p], s1);
        }
        pa[n * 32 + a]  = s0;
        pbb[n * 32 + a] = s1 + b1[a];
    }
}

// ---------------------------------------------------------------------------
// Kernel 2: pairwise MLP via WMMA. One wave owns a 16-wide j tile and loops
// over ICHUNK i values. Per i: one v_wmma_f32_16x16x32_f16 computes
// h2^T = W2 (16h x 32k, A, loop-invariant) x h1^T (32k x 16 pairs, B).
//
// Layouts used here (A/B share contiguous k order; C per CDNA5 ISA 7.12.2):
//  A (16-bit, 16x32): lanes 0-15 row M=l, elems = K 0..15;
//                     lanes 16-31 row M=l, elems = K 16..31.
//  B (16-bit, 32x16): lanes 0-15 col N=l, elems = K 0..15;
//                     lanes 16-31 col N=l, elems = K 16..31.
//  C (f32 16x16):     lane l(+16) VGPR v -> M = v (+8), N = l.
// With M=h, N=pair: lane l holds h=0..7 (half0) / h=8..15 (half1) for pair
// j0+l -> epilogue is 8 in-lane relu-FMAs + one xor-16 lane swap.
//
// pa[i] row loads are manually double-buffered so the s_wait_loadcnt for
// iteration ii+1 lands after the WMMA + epilogue of iteration ii.
// ---------------------------------------------------------------------------
__global__ __launch_bounds__(256)
void pairwise_sim_wmma(const float* __restrict__ pa,   // 2048 x 32
                       const float* __restrict__ pbb,  // 2048 x 32
                       const float* __restrict__ W2,   // 16 x 32
                       const float* __restrict__ b2,   // 16
                       const float* __restrict__ W3,   // 1 x 16
                       const float* __restrict__ b3,   // 1
                       float* __restrict__ out)        // 2048 x 2048
{
    const int tid  = threadIdx.x;
    const int wave = tid >> 5;
    const int lane = tid & 31;
    const int l    = lane & 15;
    const int half = lane >> 4;
    const int koff = half * 16;          // B-operand K half owned by this lane

    const int task = blockIdx.x * 8 + wave;   // 8 waves / block
    const int JT   = NN / 16;                 // 128 j tiles
    const int jt   = task % JT;
    const int ic   = task / JT;               // NN/ICHUNK i-chunks
    const int j0   = jt * 16;
    const int i0   = ic * ICHUNK;

    // A operand = W2, arranged per 16-bit A layout. Loop-invariant.
    v16h aw;
    #pragma unroll
    for (int e = 0; e < 16; ++e) {
        const int k = e + half * 16;
        aw[e] = (_Float16)W2[l * 32 + k];
    }

    // Epilogue constants: C VGPR v corresponds to h = v + half*8.
    float b2r[8], w3r[8];
    #pragma unroll
    for (int v = 0; v < 8; ++v) {
        b2r[v] = b2[v + half * 8];
        w3r[v] = W3[v + half * 8];
    }
    // Fold b3 into half-0's partial sum (sigmoid only evaluated on half 0).
    const float accInit = (half == 0) ? b3[0] : 0.f;

    // Hoisted per-lane pb row (+b1): pair j = j0 + l, 16 consecutive k.
    float spb[16];
    {
        const float4* pbv = (const float4*)(pbb + (j0 + l) * 32 + koff);
        #pragma unroll
        for (int q = 0; q < 4; ++q) {
            float4 t = pbv[q];
            spb[q*4+0] = t.x; spb[q*4+1] = t.y;
            spb[q*4+2] = t.z; spb[q*4+3] = t.w;
        }
    }

    // Prime the pa double buffer with i = i0.
    float4 cur[4];
    {
        const float4* pav = (const float4*)(pa + (size_t)i0 * 32 + koff);
        #pragma unroll
        for (int q = 0; q < 4; ++q) cur[q] = pav[q];
    }

    #pragma unroll 2
    for (int ii = 0; ii < ICHUNK; ++ii) {
        const int i = i0 + ii;

        // Kick off next iteration's pa row loads (clamped on last iter).
        const int inext = i0 + ((ii < ICHUNK - 1) ? ii + 1 : ICHUNK - 1);
        const float4* pnv = (const float4*)(pa + (size_t)inext * 32 + koff);
        float4 nxt0 = pnv[0];
        float4 nxt1 = pnv[1];
        float4 nxt2 = pnv[2];
        float4 nxt3 = pnv[3];

        // Build B operand: h1^T column for 16 pairs; lane supplies its 16 k's.
        v16h bm;
        #pragma unroll
        for (int q = 0; q < 4; ++q) {
            float4 t = cur[q];
            bm[q*4+0] = (_Float16)fmaxf(spb[q*4+0] + t.x, 0.f);
            bm[q*4+1] = (_Float16)fmaxf(spb[q*4+1] + t.y, 0.f);
            bm[q*4+2] = (_Float16)fmaxf(spb[q*4+2] + t.z, 0.f);
            bm[q*4+3] = (_Float16)fmaxf(spb[q*4+3] + t.w, 0.f);
        }

        // h2^T = A(W2) x B(h1^T): 16h x 16pairs in f32.
        v8f c = {};
        c = __builtin_amdgcn_wmma_f32_16x16x32_f16(
                /*neg_a=*/false, aw, /*neg_b=*/false, bm,
                /*c_mod=*/(short)0, c, /*reuse_a=*/false, /*reuse_b=*/false);

        // sum_h relu(h2+b2)*W3 (+ b3 on half 0): 8 in-lane FMAs,
        // then combine h-halves with a single xor-16 swap.
        float acc = accInit;
        #pragma unroll
        for (int v = 0; v < 8; ++v)
            acc = fmaf(fmaxf(c[v] + b2r[v], 0.f), w3r[v], acc);
        acc += __shfl_xor(acc, 16, 32);

        if (half == 0) {
            // Fast sigmoid: single v_exp + v_rcp (no IEEE divide sequence).
            float sim = __builtin_amdgcn_rcpf(1.0f + __expf(-acc));
            if (i == j0 + l) sim = 1.0f;           // diagonal forced to 1
            out[(size_t)i * NN + (j0 + l)] = sim;
        }

        cur[0] = nxt0; cur[1] = nxt1; cur[2] = nxt2; cur[3] = nxt3;
    }
}

extern "C" void kernel_launch(void* const* d_in, const int* in_sizes, int n_in,
                              void* d_out, int out_size, void* d_ws, size_t ws_size,
                              hipStream_t stream) {
    const float* asset = (const float*)d_in[0];
    const float* Wp    = (const float*)d_in[1];
    const float* bp    = (const float*)d_in[2];
    const float* W1    = (const float*)d_in[3];
    const float* b1    = (const float*)d_in[4];
    const float* W2    = (const float*)d_in[5];
    const float* b2    = (const float*)d_in[6];
    const float* W3    = (const float*)d_in[7];
    const float* b3    = (const float*)d_in[8];
    float* out = (float*)d_out;

    float* pa  = (float*)d_ws;                 // 2048*32 floats
    float* pbb = pa + NN * 32;                 // 2048*32 floats (pb + b1)

    precompute_papb<<<NN / 256, 256, 0, stream>>>(asset, Wp, bp, W1, b1, pa, pbb);

    // 128 j-tiles x (NN/ICHUNK) i-chunks = 4096 wave tasks, 8 waves / block.
    pairwise_sim_wmma<<<(128 * (NN / ICHUNK)) / 8, 256, 0, stream>>>(
        pa, pbb, W2, b2, W3, b3, out);
}
